// MoELayer_57569741635630
// MI455X (gfx1250) — matrix operand order repaired and checked by reading, not compiled, not run on hardware
//
#include <hip/hip_runtime.h>

// ---------------- problem constants ----------------
#define T_TOK 8192   // B*S tokens
#define HDIM  1024
#define FDIM  4096
#define NEXP  8
#define TOPK  2

// ---------------- FFN tiling ----------------
#define TM    32     // token rows per FFN block
#define NTHR  512    // 16 waves (wave32)
#define XS    1032   // Xtile row stride in elems (2064 B = 129*16, conflict-free, 16B aligned)
#define BS    136    // B/act tile row stride in elems (272 B = 17*16)

// gfx1250 async global->LDS DMA staging (tracked by ASYNCcnt).
// Flip to 0 to fall back to synchronous vector copies.
#define USE_ASYNC 1

typedef __attribute__((ext_vector_type(16))) __bf16        v16bf;
typedef __attribute__((ext_vector_type(8)))  float         v8f;
typedef __attribute__((ext_vector_type(8)))  unsigned int  u32x8;
typedef __attribute__((ext_vector_type(4)))  unsigned int  u32x4;

#if USE_ASYNC
#define ASYNC_WAIT() asm volatile("s_wait_asynccnt 0x0" ::: "memory")
#else
#define ASYNC_WAIT()
#endif

__device__ __forceinline__ unsigned short f2bf(float f) {
  unsigned int u = __float_as_uint(f);
  unsigned int r = u + 0x7fffu + ((u >> 16) & 1u);   // round-to-nearest-even
  return (unsigned short)(r >> 16);
}

__device__ __forceinline__ float gelu_tanh(float v) {
  const float c0 = 0.7978845608028654f;              // sqrt(2/pi)
  float u = c0 * (v + 0.044715f * v * v * v);
  return 0.5f * v * (1.0f + tanhf(u));
}

__device__ __forceinline__ v8f zero_v8f() {
  v8f z;
#pragma unroll
  for (int i = 0; i < 8; ++i) z[i] = 0.0f;
  return z;
}

// A-fragment 16x32 bf16 from LDS row-major [m][k] (ISA 7.12.2):
// lane<16: row=lane,    K = {0..7, 16..23};  lane>=16: row=lane-16, K = {8..15, 24..31}
// -> two aligned 16B LDS loads per lane.
__device__ __forceinline__ v16bf load_a_frag(const unsigned short* lds, int stride, int m0, int k0) {
  const int lane  = threadIdx.x & 31;
  const int mm    = m0 + (lane & 15);
  const int kbase = (lane < 16) ? 0 : 8;
  const unsigned short* p = lds + mm * stride + k0 + kbase;
  u32x4 lo = *(const u32x4*)p;         // K = kbase .. kbase+7
  u32x4 hi = *(const u32x4*)(p + 16);  // K = kbase+16 .. kbase+23
  u32x8 r;
#pragma unroll
  for (int i = 0; i < 4; ++i) { r[i] = lo[i]; r[i + 4] = hi[i]; }
  return __builtin_bit_cast(v16bf, r);
}

// B-fragment 32x16 bf16 from LDS stored transposed [n][k]:
// lane<16: col=lane, K = {0..7, 16..23};  lane>=16: col=lane-16, K = {8..15, 24..31}.
__device__ __forceinline__ v16bf load_b_frag(const unsigned short* lds, int stride, int n0, int k0) {
  const int lane = threadIdx.x & 31;
  const int nn   = n0 + (lane & 15);
  const int kh   = (lane < 16) ? 0 : 8;
  const unsigned short* p = lds + nn * stride + k0 + kh;
  u32x4 lo = *(const u32x4*)p;
  u32x4 hi = *(const u32x4*)(p + 16);
  u32x8 r;
#pragma unroll
  for (int i = 0; i < 4; ++i) { r[i] = lo[i]; r[i + 4] = hi[i]; }
  return __builtin_bit_cast(v16bf, r);
}

// Stage a 128(n) x 128(k) bf16 tile (rows k-contiguous in global) into LDS [n][k].
// 2048 16-byte segments, 512 threads -> 4 per thread.
__device__ __forceinline__ void stage_tile(const unsigned short* __restrict__ g, unsigned int grs,
                                           unsigned short* lds, int tid) {
#pragma unroll
  for (int i = 0; i < 4; ++i) {
    int s  = tid + i * NTHR;
    int n  = s >> 4;
    int sg = s & 15;
    unsigned short* lp = lds + n * BS + sg * 8;
#if USE_ASYNC
    unsigned int ldsoff = (unsigned int)(size_t)lp;               // low 32 bits = LDS offset
    unsigned int goff   = ((unsigned int)n * grs + (unsigned int)sg * 8u) * 2u;
    asm volatile("global_load_async_to_lds_b128 %0, %1, %2"
                 :: "v"(ldsoff), "v"(goff), "s"((unsigned long long)(size_t)g)
                 : "memory");
#else
    *(u32x4*)lp = *(const u32x4*)(g + (size_t)n * grs + sg * 8);
#endif
  }
}

// ---------------- router: logits -> softmax -> top2 -> dispatch; x -> bf16 ----------------
__global__ __launch_bounds__(256) void router_kernel(
    const float* __restrict__ x, const float* __restrict__ rw, const float* __restrict__ rb,
    unsigned short* __restrict__ xb, float* __restrict__ topw,
    int* __restrict__ counts, int* __restrict__ bucket) {
  __shared__ float rws[HDIM * NEXP];          // 32 KB
  const int tid = threadIdx.x;
  for (int l = tid; l < HDIM * NEXP; l += 256) rws[l] = rw[l];
  __syncthreads();

  const int t = blockIdx.x * 256 + tid;
  const float* xr = x + (size_t)t * HDIM;

  float acc[NEXP];
#pragma unroll
  for (int e = 0; e < NEXP; ++e) acc[e] = rb[e];

  for (int h = 0; h < HDIM; ++h) {
    float xv = xr[h];
    xb[(size_t)t * HDIM + h] = f2bf(xv);
#pragma unroll
    for (int e = 0; e < NEXP; ++e) acc[e] = fmaf(xv, rws[h * NEXP + e], acc[e]);
  }

  float mx = acc[0];
#pragma unroll
  for (int e = 1; e < NEXP; ++e) mx = fmaxf(mx, acc[e]);
  float p[NEXP];
#pragma unroll
  for (int e = 0; e < NEXP; ++e) p[e] = __expf(acc[e] - mx);

  int i0 = 0, i1 = -1;
  float p0 = p[0], p1 = -1.0f;
#pragma unroll
  for (int e = 1; e < NEXP; ++e) {
    if (p[e] > p0)      { i1 = i0; p1 = p0; i0 = e; p0 = p[e]; }
    else if (p[e] > p1) { i1 = e;  p1 = p[e]; }
  }
  float inv = 1.0f / (p0 + p1);
  topw[2 * t + 0] = p0 * inv;
  topw[2 * t + 1] = p1 * inv;

  int pos0 = atomicAdd(&counts[i0], 1);
  bucket[i0 * T_TOK + pos0] = 2 * t + 0;
  int pos1 = atomicAdd(&counts[i1], 1);
  bucket[i1 * T_TOK + pos1] = 2 * t + 1;
}

// ---------------- fp32 -> bf16 transpose (per expert): s[R][C] -> d[C][R] ----------------
__global__ __launch_bounds__(256) void transpose_bf16(const float* __restrict__ s,
                                                      unsigned short* __restrict__ d,
                                                      int R, int C) {
  __shared__ unsigned short tile[32][33];
  const float* se = s + (size_t)blockIdx.z * R * C;
  unsigned short* de = d + (size_t)blockIdx.z * R * C;
  const int c0 = blockIdx.x * 32, r0 = blockIdx.y * 32;
  const int tx = threadIdx.x & 31, ty = threadIdx.x >> 5;   // 32 x 8
#pragma unroll
  for (int i = 0; i < 32; i += 8)
    tile[ty + i][tx] = f2bf(se[(size_t)(r0 + ty + i) * C + c0 + tx]);
  __syncthreads();
#pragma unroll
  for (int i = 0; i < 32; i += 8)
    de[(size_t)(c0 + ty + i) * R + r0 + tx] = tile[tx][ty + i];
}

// ---------------- fused per-expert FFN: gather -> GEMM1 -> GELU -> GEMM2 ----------------
// w1t: [E][F][H] bf16 (k=H contiguous), w2t: [E][H][F] bf16 (k=F contiguous)
__global__ __launch_bounds__(NTHR) void moe_ffn_kernel(
    const unsigned short* __restrict__ xb,
    const unsigned short* __restrict__ w1t,
    const unsigned short* __restrict__ w2t,
    const float* __restrict__ b1, const float* __restrict__ b2,
    const int* __restrict__ counts, const int* __restrict__ bucket,
    float* __restrict__ yb) {
  const int e     = blockIdx.y;
  const int cnt   = counts[e];
  const int tile0 = blockIdx.x * TM;
  if (tile0 >= cnt) return;

  __shared__ unsigned short Xtile[TM * XS];      // 66.0 KB gathered bf16 token rows
  __shared__ unsigned short Bst[2][128 * BS];    // 69.6 KB double-buffered weight tile [n][k]
  __shared__ unsigned short actS[TM * BS];       //  8.7 KB gelu(act) tile
  __shared__ int rowtk[TM];

  const int tid  = threadIdx.x;
  const int lane = tid & 31;
  const int w    = tid >> 5;    // 0..15
  const int mt   = w & 1;       // m-tile (0..1)
  const int nt   = w >> 1;      // n-tile (0..7)

  if (tid < TM) {
    int idx = tile0 + tid;
    rowtk[tid] = (idx < cnt) ? bucket[e * T_TOK + idx] : -1;
  }
  __syncthreads();

  const unsigned short* w1te = w1t + (size_t)e * HDIM * FDIM;
  const unsigned short* w2te = w2t + (size_t)e * HDIM * FDIM;
  const float* b1e = b1 + e * FDIM;
  const float* b2e = b2 + e * HDIM;

  // gather X rows (vector copies) + prologue B-tile DMA (W1 f-chunk 0, k-stage 0)
#pragma unroll
  for (int i = 0; i < 8; ++i) {
    int s = tid + i * NTHR;
    int r = s >> 7, sg = s & 127;
    int tk = rowtk[r];
    u32x4 v = {0u, 0u, 0u, 0u};
    if (tk >= 0) v = *(const u32x4*)(xb + (size_t)(tk >> 1) * HDIM + sg * 8);
    *(u32x4*)(Xtile + r * XS + sg * 8) = v;
  }
  int buf = 0;
  stage_tile(w1te, HDIM, &Bst[0][0], tid);
  ASYNC_WAIT();
  __syncthreads();

  v8f acc2[8];
#pragma unroll
  for (int c = 0; c < 8; ++c) acc2[c] = zero_v8f();

  for (int fc = 0; fc < FDIM / 128; ++fc) {
    const int f0 = fc * 128;
    v8f acc1 = zero_v8f();

    // ---- GEMM1: act[32,128] = X[32,H] * W1[H, f0:f0+128], 8 k-stages of 128 ----
#pragma unroll
    for (int ks = 0; ks < 8; ++ks) {
      if (ks < 7)
        stage_tile(w1te + (size_t)f0 * HDIM + (ks + 1) * 128, HDIM, &Bst[buf ^ 1][0], tid);
      else
        stage_tile(w2te + (size_t)f0, FDIM, &Bst[buf ^ 1][0], tid);   // GEMM2 c=0 tile
      const unsigned short* Bcur = &Bst[buf][0];
#pragma unroll
      for (int kk = 0; kk < 4; ++kk) {
        v16bf a = load_a_frag(Xtile, XS, mt * 16, ks * 128 + kk * 32);
        v16bf b = load_b_frag(Bcur, BS, nt * 16, kk * 32);
        acc1 = __builtin_amdgcn_wmma_f32_16x16x32_bf16(false, a, false, b,
                                                       (short)0, acc1, false, false);
      }
      ASYNC_WAIT();
      __syncthreads();
      buf ^= 1;
    }

    // ---- bias + GELU -> bf16 act tile ----
    {
      const int nc = nt * 16 + (lane & 15);
      const float bias = b1e[f0 + nc];
#pragma unroll
      for (int r = 0; r < 8; ++r) {
        int m = mt * 16 + r + ((lane & 16) ? 8 : 0);
        actS[m * BS + nc] = f2bf(gelu_tanh(acc1[r] + bias));
      }
    }
    if (fc + 1 < FDIM / 128)
      __builtin_prefetch(w1te + (size_t)(f0 + 128) * HDIM + (size_t)tid * 32, 0, 1);
    __syncthreads();

    // ---- GEMM2: out[32,1024] += act[32,128] * W2[f0:f0+128, :], 8 n-chunks of 128 ----
#pragma unroll
    for (int c = 0; c < 8; ++c) {
      if (c < 7)
        stage_tile(w2te + (size_t)((c + 1) * 128) * FDIM + f0, FDIM, &Bst[buf ^ 1][0], tid);
      else if (fc + 1 < FDIM / 128)
        stage_tile(w1te + (size_t)(f0 + 128) * HDIM, HDIM, &Bst[buf ^ 1][0], tid);
      const unsigned short* Bcur = &Bst[buf][0];
#pragma unroll
      for (int kk = 0; kk < 4; ++kk) {
        v16bf a = load_a_frag(actS, BS, mt * 16, kk * 32);
        v16bf b = load_b_frag(Bcur, BS, nt * 16, kk * 32);
        acc2[c] = __builtin_amdgcn_wmma_f32_16x16x32_bf16(false, a, false, b,
                                                          (short)0, acc2[c], false, false);
      }
      ASYNC_WAIT();
      __syncthreads();
      buf ^= 1;
    }
  }

  // ---- epilogue: +b2, scatter rows to per-(token,k) slots ----
#pragma unroll
  for (int c = 0; c < 8; ++c) {
    const int n = c * 128 + nt * 16 + (lane & 15);
    const float bias2 = b2e[n];
#pragma unroll
    for (int r = 0; r < 8; ++r) {
      int m = mt * 16 + r + ((lane & 16) ? 8 : 0);
      int tk = rowtk[m];
      if (tk >= 0) yb[(size_t)tk * HDIM + n] = acc2[c][r] + bias2;
    }
  }
}

// ---------------- combine: out[t] = w0*y(t,0) + w1*y(t,1) ----------------
__global__ __launch_bounds__(256) void combine_kernel(
    const float* __restrict__ yb, const float* __restrict__ topw, float* __restrict__ out) {
  int idx = blockIdx.x * 256 + threadIdx.x;   // over T*(H/4)
  int t  = idx >> 8;
  int h0 = (idx & 255) << 2;
  float w0 = topw[2 * t + 0];
  float w1 = topw[2 * t + 1];
  const float* a = yb + (size_t)(2 * t + 0) * HDIM + h0;
  const float* b = yb + (size_t)(2 * t + 1) * HDIM + h0;
  float* o = out + (size_t)t * HDIM + h0;
#pragma unroll
  for (int j = 0; j < 4; ++j) o[j] = w0 * a[j] + w1 * b[j];
}

extern "C" void kernel_launch(void* const* d_in, const int* in_sizes, int n_in,
                              void* d_out, int out_size, void* d_ws, size_t ws_size,
                              hipStream_t stream) {
  const float* x  = (const float*)d_in[0];   // [B,S,H]
  const float* w1 = (const float*)d_in[1];   // [E,H,F]
  const float* b1 = (const float*)d_in[2];   // [E,F]
  const float* w2 = (const float*)d_in[3];   // [E,F,H]
  const float* b2 = (const float*)d_in[4];   // [E,H]
  const float* rw = (const float*)d_in[5];   // [H,E]
  const float* rb = (const float*)d_in[6];   // [E]
  float* out = (float*)d_out;

  char* ws = (char*)d_ws;
  size_t off = 0;
  unsigned short* xb  = (unsigned short*)(ws + off); off += (size_t)T_TOK * HDIM * 2;        // 16 MB
  unsigned short* w1t = (unsigned short*)(ws + off); off += (size_t)NEXP * HDIM * FDIM * 2;  // 64 MB
  unsigned short* w2t = (unsigned short*)(ws + off); off += (size_t)NEXP * HDIM * FDIM * 2;  // 64 MB
  float* yb           = (float*)(ws + off);          off += (size_t)T_TOK * TOPK * HDIM * 4; // 64 MB
  float* topw         = (float*)(ws + off);          off += (size_t)T_TOK * TOPK * 4;
  int* counts         = (int*)(ws + off);            off += 256;
  int* bucket         = (int*)(ws + off);            off += (size_t)NEXP * T_TOK * 4;

  hipMemsetAsync(counts, 0, 256, stream);

  router_kernel<<<T_TOK / 256, 256, 0, stream>>>(x, rw, rb, xb, topw, counts, bucket);

  // w1 [H,F] -> w1t [F,H];  w2 [F,H] -> w2t [H,F]  (bf16, k-dim contiguous)
  transpose_bf16<<<dim3(FDIM / 32, HDIM / 32, NEXP), 256, 0, stream>>>(w1, w1t, HDIM, FDIM);
  transpose_bf16<<<dim3(HDIM / 32, FDIM / 32, NEXP), 256, 0, stream>>>(w2, w2t, FDIM, HDIM);

  dim3 grid(T_TOK / TM, NEXP);
  moe_ffn_kernel<<<grid, NTHR, 0, stream>>>(xb, w1t, w2t, b1, b2, counts, bucket, yb);

  combine_kernel<<<(T_TOK * (HDIM / 4)) / 256, 256, 0, stream>>>(yb, topw, out);
}
